// MLPConv2d_7352984010772
// MI455X (gfx1250) — matrix operand order repaired and checked by reading, not compiled, not run on hardware
//
#include <hip/hip_runtime.h>
#include <hip/hip_bf16.h>

typedef __attribute__((ext_vector_type(16))) __bf16 v16bf;
typedef __attribute__((ext_vector_type(8)))  float  v8f;

union FragBF {
    v16bf v;
    uint4 u[2];
};

#define H_IMG 224
#define W_IMG 224
#define C_INC 64
#define C_OUTC 64
#define HIDC 64
#define IN_SZ 576

// LDS slab: [row(18)][col(18)][chan padded to 72] bf16 -> 46656 B
#define SLAB_CP 72
// per-wave H scratch: [m(32)][hid padded to 72] bf16, 8 waves -> 36864 B
#define HP 72

// Slab (conv1 input stage) and H scratch (conv1->conv2 bounce) are never
// live at the same time; union them to stay under the static-LDS limit.
union SharedMem {
    __bf16 slab[18 * 18 * SLAB_CP];
    __bf16 hbuf[8 * 32 * HP];
};

// ws layout (bf16 elements):
//   [0, 36864)      W1 packed: [kstep(18)][ntile(4)][n(16)][k(32)]
//   [36864, 40960)  W2 packed: [kstep(2)][ntile(4)][n(16)][k(32)]
#define W1P_ELEMS 36864
#define W2P_ELEMS 4096

__global__ __launch_bounds__(256)
void repack_weights_kernel(const float* __restrict__ W1,
                           const float* __restrict__ W2,
                           __bf16* __restrict__ wsp)
{
    int i = blockIdx.x * 256 + threadIdx.x;
    if (i < W1P_ELEMS) {
        int kstep = i >> 11;          // / 2048
        int rem   = i & 2047;
        int t     = rem >> 9;         // ntile
        int n     = (rem >> 5) & 15;
        int kk    = rem & 31;
        int tap   = kstep >> 1;
        int c     = ((kstep & 1) << 5) + kk;
        int f     = c * 9 + tap;      // unfold feature order (c, kh, kw)
        int o     = t * 16 + n;
        wsp[i] = (__bf16)W1[o * IN_SZ + f];
    } else if (i < W1P_ELEMS + W2P_ELEMS) {
        int j     = i - W1P_ELEMS;
        int kstep = j >> 11;
        int rem   = j & 2047;
        int t     = rem >> 9;
        int n     = (rem >> 5) & 15;
        int kk    = rem & 31;
        int hid   = kstep * 32 + kk;
        int o     = t * 16 + n;
        wsp[i] = (__bf16)W2[o * HIDC + hid];
    }
}

__global__ __launch_bounds__(256)
void fused_mlpconv_kernel(const float* __restrict__ x,
                          const float* __restrict__ b1,
                          const float* __restrict__ b2,
                          const __bf16* __restrict__ w1p,
                          const __bf16* __restrict__ w2p,
                          float* __restrict__ out)
{
    __shared__ SharedMem sm;

    const int tid  = threadIdx.x;
    const int wave = tid >> 5;
    const int lane = tid & 31;
    const int l16  = lane & 15;
    const int hi   = (lane >> 4) & 1;    // 0: lanes 0-15, 1: lanes 16-31

    const int w0 = blockIdx.x * 16;
    const int h0 = blockIdx.y * 16;
    const int b  = blockIdx.z;

    // ---- phase 1: stage halo slab (rows h0-1..h0+16, cols w0-1..w0+16, 64 ch)
    for (int p = wave; p < 18 * 64; p += 8) {
        int r = p >> 6;                  // slab row 0..17
        int c = p & 63;                  // channel
        if (lane < 18) {
            int h = h0 + r - 1;
            int w = w0 + lane - 1;
            float v = 0.0f;
            if (h >= 0 && h < H_IMG && w >= 0 && w < W_IMG)
                v = x[(((size_t)b * C_INC + c) * H_IMG + h) * W_IMG + w];
            sm.slab[(r * 18 + lane) * SLAB_CP + c] = (__bf16)v;
        }
    }
    __syncthreads();

    // ---- biases per lane (column o = t*16 + l16)
    float bias1[4], bias2[4];
#pragma unroll
    for (int t = 0; t < 4; ++t) {
        bias1[t] = b1[t * 16 + l16];
        bias2[t] = b2[t * 16 + l16];
    }

    // ---- conv1: each wave owns 2 M-tiles (image rows h0+2*wave, h0+2*wave+1)
    v8f acc1[2][4];
#pragma unroll
    for (int i = 0; i < 2; ++i)
#pragma unroll
        for (int t = 0; t < 4; ++t) acc1[i][t] = 0.0f;

    for (int tap = 0; tap < 9; ++tap) {
        const int dh = tap / 3;          // 0..2  (== kh)
        const int dw = tap % 3;          // 0..2  (== kw)
        const int cs = l16 + dw;         // slab col for m = l16
        const __bf16* ab0 = &sm.slab[(((2 * wave + dh) * 18) + cs) * SLAB_CP];
        const __bf16* ab1 = ab0 + 18 * SLAB_CP;   // next image row
#pragma unroll
        for (int kc = 0; kc < 2; ++kc) {
            const int kstep = tap * 2 + kc;
            // A fragments: per-lane K halves {0..7,16..23} / {8..15,24..31}
            const int ch0 = kc * 32 + hi * 8;
            FragBF a0, a1;
            a0.u[0] = *(const uint4*)(ab0 + ch0);
            a0.u[1] = *(const uint4*)(ab0 + ch0 + 16);
            a1.u[0] = *(const uint4*)(ab1 + ch0);
            a1.u[1] = *(const uint4*)(ab1 + ch0 + 16);
            const __bf16* bbase = w1p + (size_t)kstep * 2048;
#pragma unroll
            for (int t = 0; t < 4; ++t) {
                FragBF bf;
                const __bf16* bp = bbase + (t * 16 + l16) * 32 + hi * 8;
                bf.u[0] = *(const uint4*)(bp);
                bf.u[1] = *(const uint4*)(bp + 16);
                acc1[0][t] = __builtin_amdgcn_wmma_f32_16x16x32_bf16(
                    false, a0.v, false, bf.v, (short)0, acc1[0][t], false, false);
                acc1[1][t] = __builtin_amdgcn_wmma_f32_16x16x32_bf16(
                    false, a1.v, false, bf.v, (short)0, acc1[1][t], false, false);
            }
        }
    }

    // All slab reads must finish before hbuf overwrites the same LDS.
    __syncthreads();

    // ---- bias + ReLU, reshape C-layout -> A-layout via per-wave LDS scratch
    __bf16* hb = &sm.hbuf[wave * 32 * HP];
#pragma unroll
    for (int i = 0; i < 2; ++i) {
#pragma unroll
        for (int t = 0; t < 4; ++t) {
#pragma unroll
            for (int g = 0; g < 8; ++g) {
                float v = acc1[i][t][g] + bias1[t];
                v = v > 0.0f ? v : 0.0f;
                int m = i * 16 + g + hi * 8;
                hb[m * HP + t * 16 + l16] = (__bf16)v;
            }
        }
    }

    // ---- 1x1 conv (K = 64 = 2 k-chunks), 2 M-tiles per wave
    v8f acc2[2][4];
#pragma unroll
    for (int i = 0; i < 2; ++i)
#pragma unroll
        for (int t = 0; t < 4; ++t) acc2[i][t] = 0.0f;

#pragma unroll
    for (int kc = 0; kc < 2; ++kc) {
        const int ch0 = kc * 32 + hi * 8;
        FragBF a0, a1;
        const __bf16* ab0 = hb + l16 * HP + ch0;
        const __bf16* ab1 = hb + (16 + l16) * HP + ch0;
        a0.u[0] = *(const uint4*)(ab0);
        a0.u[1] = *(const uint4*)(ab0 + 16);
        a1.u[0] = *(const uint4*)(ab1);
        a1.u[1] = *(const uint4*)(ab1 + 16);
        const __bf16* bbase = w2p + (size_t)kc * 2048;
#pragma unroll
        for (int t = 0; t < 4; ++t) {
            FragBF bf;
            const __bf16* bp = bbase + (t * 16 + l16) * 32 + hi * 8;
            bf.u[0] = *(const uint4*)(bp);
            bf.u[1] = *(const uint4*)(bp + 16);
            acc2[0][t] = __builtin_amdgcn_wmma_f32_16x16x32_bf16(
                false, a0.v, false, bf.v, (short)0, acc2[0][t], false, false);
            acc2[1][t] = __builtin_amdgcn_wmma_f32_16x16x32_bf16(
                false, a1.v, false, bf.v, (short)0, acc2[1][t], false, false);
        }
    }

    // ---- bias + store f32
#pragma unroll
    for (int i = 0; i < 2; ++i) {
        const int hrow = h0 + 2 * wave + i;
#pragma unroll
        for (int t = 0; t < 4; ++t) {
            const int o = t * 16 + l16;
            float* op = out + (((size_t)b * C_OUTC + o) * H_IMG + hrow) * W_IMG + w0;
#pragma unroll
            for (int g = 0; g < 8; ++g) {
                op[g + hi * 8] = acc2[i][t][g] + bias2[t];
            }
        }
    }
}

extern "C" void kernel_launch(void* const* d_in, const int* in_sizes, int n_in,
                              void* d_out, int out_size, void* d_ws, size_t ws_size,
                              hipStream_t stream) {
    const float* x  = (const float*)d_in[0];
    const float* W1 = (const float*)d_in[1];
    const float* b1 = (const float*)d_in[2];
    const float* W2 = (const float*)d_in[3];
    const float* b2 = (const float*)d_in[4];
    float* out = (float*)d_out;

    __bf16* wsp = (__bf16*)d_ws;
    const __bf16* w1p = wsp;
    const __bf16* w2p = wsp + W1P_ELEMS;

    // 1) pack W1/W2 to bf16 in exact WMMA B-fragment order
    repack_weights_kernel<<<(W1P_ELEMS + W2P_ELEMS + 255) / 256, 256, 0, stream>>>(
        W1, W2, wsp);

    // 2) fused conv3x3 + bias + ReLU + conv1x1 + bias
    dim3 grid(W_IMG / 16, H_IMG / 16, 8);
    fused_mlpconv_kernel<<<grid, 256, 0, stream>>>(x, b1, b2, w1p, w2p, out);
}
